// BiLSTM_73074573574724
// MI455X (gfx1250) — matrix operand order repaired and checked by reading, not compiled, not run on hardware
//
#include <hip/hip_runtime.h>

// ============================================================================
// Stacked bi-LSTM for MI455X (gfx1250) — round 3.
//  - Templated layer shapes -> fully unrolled WMMA chains, scalar (s_cbranch)
//    tile guards via readfirstlane (no exec-mask churn around v_wmma).
//  - Activations are produced in WMMA A-fragment order; per-step input tiles
//    are contiguous blocks moved by the Tensor Data Mover (tensor_load_to_lds,
//    TENSORcnt), double-buffered to overlap DMA with the recurrent GEMM.
//  - Weights (bf16, frag-major) resident in LDS; h/c state resident in LDS.
// ============================================================================

typedef __attribute__((ext_vector_type(16))) __bf16 v16bf;
typedef __attribute__((ext_vector_type(8)))  float  v8f;
typedef __attribute__((ext_vector_type(4)))  unsigned int v4u;
typedef __attribute__((ext_vector_type(8)))  int v8i;
typedef __attribute__((ext_vector_type(4)))  int v4i;

#define SEQ     2048
#define BATCH   128
#define NTILE_B 8      // batch tiles of 16
#define F_OUT   40

__device__ __forceinline__ float sigmoidf_(float x) {
  return 1.0f / (1.0f + __expf(-x));
}

// A-matrix (16x32 bf16) fragment layout per CDNA5 ISA 7.12.2
__device__ __forceinline__ int a_idx(int m, int k) {
  int kt = k >> 5, kk = k & 31;
  int hi = (kk >> 3) & 1;
  int lane = m + (hi ? 16 : 0);
  int h = kk - (hi ? 8 : 0) - ((kk & 16) ? 8 : 0);
  return kt * 512 + lane * 16 + h;
}
// B-matrix (32x16 bf16) fragment layout
__device__ __forceinline__ int b_idx(int k, int n, int NT) {
  int kt = k >> 5, kk = k & 31, nt = n >> 4, nn = n & 15;
  return ((kt * NT) + nt) * 512 + (nn + (((kk >> 3) & 1) ? 16 : 0)) * 16 + ((kk & 7) + ((kk & 16) >> 1));
}

// ---------------------------------------------------------------------------
// TDM: 1-D contiguous global -> LDS copy of nqw 8-byte words.
// D# group0: count=1, lds_addr, 57-bit global addr, type=2 ("image").
// D# group1: data_size=8B, tensor_dim0=tile_dim0=nqw (1-D tile).
// This toolchain's builtin takes 6 args (g0, g1, g2, g3, g4, cpol).
// ---------------------------------------------------------------------------
__device__ __forceinline__ void tdm_load_1d(unsigned ldsOff, const void* g,
                                            unsigned nqw) {
  unsigned long long ga = (unsigned long long)(uintptr_t)g;
  unsigned d0 = 1u;                                            // count = 1
  unsigned d1 = ldsOff;                                        // lds_addr
  unsigned d2 = (unsigned)ga;                                  // ga[31:0]
  unsigned d3 = ((unsigned)(ga >> 32) & 0x01FFFFFFu) | 0x80000000u; // +type=2
  unsigned e0 = 3u << 16;                                      // data_size=8B
  unsigned e1 = (nqw & 0xFFFFu) << 16;                         // dim0 lo
  unsigned e2 = ((nqw >> 16) & 0xFFFFu) | (1u << 16);          // dim0 hi|dim1=1
  unsigned e3 = (nqw & 0xFFFFu) << 16;                         // tile_dim0
  unsigned e5 = nqw;                                           // dim0 stride
  v4u g0 = { (unsigned)__builtin_amdgcn_readfirstlane((int)d0),
             (unsigned)__builtin_amdgcn_readfirstlane((int)d1),
             (unsigned)__builtin_amdgcn_readfirstlane((int)d2),
             (unsigned)__builtin_amdgcn_readfirstlane((int)d3) };
  v8i g1 = { __builtin_amdgcn_readfirstlane((int)e0),
             __builtin_amdgcn_readfirstlane((int)e1),
             __builtin_amdgcn_readfirstlane((int)e2),
             __builtin_amdgcn_readfirstlane((int)e3),
             0,
             __builtin_amdgcn_readfirstlane((int)e5),
             0, 0 };
  v4i z4 = {0, 0, 0, 0};
  v8i z8 = {0, 0, 0, 0, 0, 0, 0, 0};
  __builtin_amdgcn_tensor_load_to_lds(g0, g1, z4, z4, z8, 0);
}

// ---------------------------------------------------------------------------
// Weight swizzle: Wcat columns k: [xoff,xoff+I) <- Wih, [hoff,hoff+H) <- Whh
// stored in B-fragment-major bf16; bias fused (bih + bhh).
// ---------------------------------------------------------------------------
__global__ void prep_weights(const float* __restrict__ wih,
                             const float* __restrict__ whh,
                             const float* __restrict__ bih,
                             const float* __restrict__ bhh,
                             int I, int H, int xoff, int hoff, int Kpad, int NT,
                             __bf16* __restrict__ Wt, float* __restrict__ bias) {
  int NPAD = NT * 16;
  int total = Kpad * NPAD;
  int G = 4 * H;
  for (int e = blockIdx.x * blockDim.x + threadIdx.x; e < total;
       e += gridDim.x * blockDim.x) {
    int k = e / NPAD, n = e % NPAD;
    float v = 0.0f;
    if (n < G) {
      if (k >= xoff && k < xoff + I)       v = wih[n * I + (k - xoff)];
      else if (k >= hoff && k < hoff + H)  v = whh[n * H + (k - hoff)];
    }
    Wt[b_idx(k, n, NT)] = (__bf16)v;
  }
  for (int n = blockIdx.x * blockDim.x + threadIdx.x; n < NPAD;
       n += gridDim.x * blockDim.x)
    bias[n] = (n < G) ? (bih[n] + bhh[n]) : 0.0f;
}

__global__ void zero_u4(uint4* __restrict__ p, long long n16) {
  long long i = (long long)blockIdx.x * blockDim.x + threadIdx.x;
  long long st = (long long)gridDim.x * blockDim.x;
  uint4 z = {0u, 0u, 0u, 0u};
  for (; i < n16; i += st) p[i] = z;
}

// ---------------------------------------------------------------------------
// Layer 0: I=3 raw f32 input packed at K=[100,103) after h (K=[0,100)).
// grid = (8 batch tiles, 2 dirs), block = 256 (8 waves).
// ---------------------------------------------------------------------------
__global__ __launch_bounds__(256)
void lstm_layer0(const float* __restrict__ x, const __bf16* __restrict__ WtAll,
                 const float* __restrict__ biasAll, __bf16* __restrict__ stOut,
                 int ixtNext) {
  constexpr int H = 100, NT = 25, KT = 4, NPAD = 400, XOFF = 100;
  __shared__ __attribute__((aligned(32))) __bf16 sW[KT * NT * 512];
  __shared__ __attribute__((aligned(32))) __bf16 sA[KT * 512];
  __shared__ float sGates[16 * NPAD];
  __shared__ float sC[16 * H];
  __shared__ float sBias[NPAD];

  const int tid = threadIdx.x;
  const int lane = tid & 31;
  const int wave = __builtin_amdgcn_readfirstlane(tid >> 5);
  const int dir = blockIdx.y;
  const int tileB = blockIdx.x;
  const int b0 = tileB * 16;

  const __bf16* Wt = WtAll + (size_t)dir * 65536;
  const float* bias = biasAll + (size_t)dir * 512;
  {
    int n16 = (KT * NT * 512) >> 3;
    const uint4* src = (const uint4*)Wt;
    uint4* dst = (uint4*)sW;
    for (int i = tid; i < n16; i += 256) dst[i] = src[i];
  }
  for (int i = tid; i < KT * 512; i += 256) sA[i] = (__bf16)0.0f;
  for (int i = tid; i < 16 * H; i += 256) sC[i] = 0.0f;
  for (int i = tid; i < NPAD; i += 256) sBias[i] = bias[i];
  __syncthreads();

  for (int step = 0; step < SEQ; ++step) {
    const int s = dir ? (SEQ - 1 - step) : step;
    if (tid < 48) {  // x_t is 16x3
      int m = tid / 3, k = tid - 3 * m;
      sA[a_idx(m, XOFF + k)] = (__bf16)x[((size_t)s * BATCH + b0 + m) * 3 + k];
    }
    if (step + 1 < SEQ) {
      int s2 = dir ? (SEQ - 2 - step) : (step + 1);
      __builtin_prefetch(x + ((size_t)s2 * BATCH + b0) * 3, 0, 3);
    }
    __syncthreads();

    const int row = (lane >> 4) * 8, col = lane & 15;
    for (int t = wave; t < NT; t += 8) {        // scalar-uniform loop
      v8f acc = {};
#pragma unroll
      for (int kt = 0; kt < KT; ++kt) {
        v16bf a = *(const v16bf*)&sA[kt * 512 + lane * 16];
        v16bf b = *(const v16bf*)&sW[(kt * NT + t) * 512 + lane * 16];
        acc = __builtin_amdgcn_wmma_f32_16x16x32_bf16(
            false, a, false, b, (short)0, acc, false, false);
      }
#pragma unroll
      for (int v = 0; v < 8; ++v)
        sGates[(row + v) * NPAD + t * 16 + col] = acc[v];
    }
    __syncthreads();

    for (int e = tid; e < 16 * H; e += 256) {
      int m = e / H, j = e % H;
      float gi = sGates[m * NPAD + j]         + sBias[j];
      float gf = sGates[m * NPAD + H + j]     + sBias[H + j];
      float gg = sGates[m * NPAD + 2 * H + j] + sBias[2 * H + j];
      float go = sGates[m * NPAD + 3 * H + j] + sBias[3 * H + j];
      float c = sigmoidf_(gf) * sC[e] + sigmoidf_(gi) * tanhf(gg);
      float h = sigmoidf_(go) * tanhf(c);
      sC[e] = c;
      sA[a_idx(m, j)] = (__bf16)h;               // h at K=[0,H)
      stOut[((size_t)s * NTILE_B + tileB) * ((size_t)ixtNext * 512) +
            a_idx(m, dir * H + j)] = (__bf16)h;  // staged for next layer
    }
    __syncthreads();
  }
}

// ---------------------------------------------------------------------------
// Layers 1..3: staged fragment-order input, TDM double-buffered x tiles.
// x at K=[0,IPAD), h at K=[IPAD,IPAD+H).
// ---------------------------------------------------------------------------
template <int IPAD, int H, int NT, bool FINAL>
__global__ __launch_bounds__(256)
void lstm_tdm(const __bf16* __restrict__ stIn, const __bf16* __restrict__ WtAll,
              const float* __restrict__ biasAll, __bf16* __restrict__ stOut,
              float* __restrict__ finalOut, int ixtNext) {
  constexpr int IXT = IPAD / 32;
  constexpr int KT = (IPAD + H + 31) / 32;
  constexpr int HKT = KT - IXT;
  constexpr int NPAD = NT * 16;
  constexpr unsigned BLKH = IXT * 512;       // halves per staged block
  constexpr unsigned BLKQW = BLKH / 4;       // 8-byte words per block

  __shared__ __attribute__((aligned(32))) __bf16 sW[KT * NT * 512];
  __shared__ __attribute__((aligned(32))) __bf16 sX[2 * BLKH];
  __shared__ __attribute__((aligned(32))) __bf16 sAh[HKT * 512];
  __shared__ float sGates[16 * NPAD];
  __shared__ float sC[16 * H];
  __shared__ float sBias[NPAD];

  const int tid = threadIdx.x;
  const int lane = tid & 31;
  const int wave = __builtin_amdgcn_readfirstlane(tid >> 5);
  const int dir = blockIdx.y;
  const int tileB = blockIdx.x;
  const int b0 = tileB * 16;

  const __bf16* Wt = WtAll + (size_t)dir * 65536;
  const float* bias = biasAll + (size_t)dir * 512;
  {
    int n16 = (KT * NT * 512) >> 3;
    const uint4* src = (const uint4*)Wt;
    uint4* dst = (uint4*)sW;
    for (int i = tid; i < n16; i += 256) dst[i] = src[i];
  }
  for (int i = tid; i < HKT * 512; i += 256) sAh[i] = (__bf16)0.0f;
  for (int i = tid; i < 16 * H; i += 256) sC[i] = 0.0f;
  for (int i = tid; i < NPAD; i += 256) sBias[i] = bias[i];

  const unsigned sXoff = (unsigned)(uintptr_t)&sX[0];  // LDS byte offset
  if (wave == 0) {  // preload x for step 0 into buffer 0
    int s0 = dir ? (SEQ - 1) : 0;
    tdm_load_1d(sXoff, stIn + ((size_t)s0 * NTILE_B + tileB) * BLKH, BLKQW);
  }
  __syncthreads();

  for (int step = 0; step < SEQ; ++step) {
    const int s = dir ? (SEQ - 1 - step) : step;
    if (wave == 0) {
      __builtin_amdgcn_s_wait_tensorcnt(0);          // x[step&1] landed
      if (step + 1 < SEQ) {                          // overlap next DMA
        int s2 = dir ? (SEQ - 2 - step) : (step + 1);
        tdm_load_1d(sXoff + (unsigned)(((step + 1) & 1) * BLKH * 2),
                    stIn + ((size_t)s2 * NTILE_B + tileB) * BLKH, BLKQW);
      }
    }
    __syncthreads();

    const __bf16* sXcur = &sX[(step & 1) * BLKH];
    const int row = (lane >> 4) * 8, col = lane & 15;
    for (int t = wave; t < NT; t += 8) {             // scalar-uniform loop
      v8f acc = {};
#pragma unroll
      for (int kt = 0; kt < KT; ++kt) {
        const __bf16* asrc = (kt < IXT) ? (sXcur + kt * 512 + lane * 16)
                                        : (&sAh[(kt - IXT) * 512 + lane * 16]);
        v16bf a = *(const v16bf*)asrc;
        v16bf b = *(const v16bf*)&sW[(kt * NT + t) * 512 + lane * 16];
        acc = __builtin_amdgcn_wmma_f32_16x16x32_bf16(
            false, a, false, b, (short)0, acc, false, false);
      }
#pragma unroll
      for (int v = 0; v < 8; ++v)
        sGates[(row + v) * NPAD + t * 16 + col] = acc[v];
    }
    __syncthreads();

    for (int e = tid; e < 16 * H; e += 256) {
      int m = e / H, j = e % H;
      float gi = sGates[m * NPAD + j]         + sBias[j];
      float gf = sGates[m * NPAD + H + j]     + sBias[H + j];
      float gg = sGates[m * NPAD + 2 * H + j] + sBias[2 * H + j];
      float go = sGates[m * NPAD + 3 * H + j] + sBias[3 * H + j];
      float c = sigmoidf_(gf) * sC[e] + sigmoidf_(gi) * tanhf(gg);
      float h = sigmoidf_(go) * tanhf(c);
      sC[e] = c;
      sAh[a_idx(m, IPAD + j) - IXT * 512] = (__bf16)h;
      if constexpr (FINAL) {
        finalOut[((size_t)s * BATCH + b0 + m) * (2 * H) + (size_t)dir * H + j] = h;
      } else {
        stOut[((size_t)s * NTILE_B + tileB) * ((size_t)ixtNext * 512) +
              a_idx(m, dir * H + j)] = (__bf16)h;
      }
    }
    __syncthreads();
  }
}

// ---------------------------------------------------------------------------
// softmax over SEQ per (b,f) column, >= 0.5 threshold
// ---------------------------------------------------------------------------
__global__ void softmax_thresh(const float* __restrict__ in,
                               float* __restrict__ out) {
  __shared__ float red[256];
  const int col = blockIdx.x;
  const size_t stride = (size_t)BATCH * F_OUT;
  const size_t base = (size_t)(col / F_OUT) * F_OUT + (col % F_OUT);
  const int tid = threadIdx.x;

  float mx = -1e30f;
  for (int s = tid; s < SEQ; s += 256)
    mx = fmaxf(mx, in[base + (size_t)s * stride]);
  red[tid] = mx; __syncthreads();
  for (int o = 128; o > 0; o >>= 1) {
    if (tid < o) red[tid] = fmaxf(red[tid], red[tid + o]);
    __syncthreads();
  }
  mx = red[0]; __syncthreads();

  float sum = 0.0f;
  for (int s = tid; s < SEQ; s += 256)
    sum += __expf(in[base + (size_t)s * stride] - mx);
  red[tid] = sum; __syncthreads();
  for (int o = 128; o > 0; o >>= 1) {
    if (tid < o) red[tid] += red[tid + o];
    __syncthreads();
  }
  sum = red[0];

  const float thresh = 0.5f * sum;
  for (int s = tid; s < SEQ; s += 256) {
    float e = __expf(in[base + (size_t)s * stride] - mx);
    out[base + (size_t)s * stride] = (e >= thresh) ? 1.0f : 0.0f;
  }
}

// ---------------------------------------------------------------------------
// host entry
// ---------------------------------------------------------------------------
extern "C" void kernel_launch(void* const* d_in, const int* in_sizes, int n_in,
                              void* d_out, int out_size, void* d_ws,
                              size_t ws_size, hipStream_t stream) {
  (void)in_sizes; (void)n_in; (void)out_size; (void)ws_size;
  const float* x = (const float*)d_in[0];

  char* ws = (char*)d_ws;
  size_t off = 0;
  auto carve = [&](size_t bytes) -> void* {
    void* p = ws + off;
    off += (bytes + 255) & ~(size_t)255;
    return p;
  };

  // staged activation blocks (frag-order, zero-padded): keep st1..st3 adjacent
  const size_t ST1_H = (size_t)SEQ * NTILE_B * 7 * 512;  // L0 out (IPAD=224)
  const size_t ST2_H = (size_t)SEQ * NTILE_B * 4 * 512;  // L1 out (IPAD=128)
  const size_t ST3_H = (size_t)SEQ * NTILE_B * 4 * 512;  // L2 out (IPAD=128)
  __bf16* st1 = (__bf16*)carve(ST1_H * 2);
  __bf16* st2 = (__bf16*)carve(ST2_H * 2);
  __bf16* st3 = (__bf16*)carve(ST3_H * 2);
  float* fin = (float*)carve((size_t)SEQ * BATCH * F_OUT * sizeof(float));
  __bf16* wt[4]; float* bi[4];
  for (int L = 0; L < 4; ++L) {
    wt[L] = (__bf16*)carve((size_t)2 * 65536 * 2);
    bi[L] = (float*)carve((size_t)2 * 512 * 4);
  }

  // layer configs: {I, H, xoff, hoff, Kpad, NT}
  const int cfg[4][6] = {
      {3,   100, 100, 0,   128, 25},   // L0: h at K=[0,100), x at [100,103)
      {200, 50,  0,   224, 288, 13},   // L1: x at [0,224), h at [224,274)
      {100, 50,  0,   128, 192, 13},   // L2
      {100, 20,  0,   128, 160, 5},    // L3
  };
  for (int L = 0; L < 4; ++L) {
    const int base = 1 + 8 * L;
    int total = cfg[L][4] * cfg[L][5] * 16;
    int blocks = (total + 255) / 256;
    prep_weights<<<blocks, 256, 0, stream>>>(
        (const float*)d_in[base + 0], (const float*)d_in[base + 1],
        (const float*)d_in[base + 2], (const float*)d_in[base + 3],
        cfg[L][0], cfg[L][1], cfg[L][2], cfg[L][3], cfg[L][4], cfg[L][5],
        wt[L], bi[L]);
    prep_weights<<<blocks, 256, 0, stream>>>(
        (const float*)d_in[base + 4], (const float*)d_in[base + 5],
        (const float*)d_in[base + 6], (const float*)d_in[base + 7],
        cfg[L][0], cfg[L][1], cfg[L][2], cfg[L][3], cfg[L][4], cfg[L][5],
        wt[L] + 65536, bi[L] + 512);
  }

  // zero staged buffers once (pad lanes must read as 0.0)
  long long zb16 = (long long)((ST1_H + ST2_H + ST3_H) * 2 / 16);
  zero_u4<<<4096, 256, 0, stream>>>((uint4*)st1, zb16);

  dim3 grid(NTILE_B, 2);
  lstm_layer0<<<grid, 256, 0, stream>>>(x, wt[0], bi[0], st1, 7);
  lstm_tdm<224, 50, 13, false><<<grid, 256, 0, stream>>>(st1, wt[1], bi[1],
                                                         st2, nullptr, 4);
  lstm_tdm<128, 50, 13, false><<<grid, 256, 0, stream>>>(st2, wt[2], bi[2],
                                                         st3, nullptr, 4);
  lstm_tdm<128, 20, 5, true><<<grid, 256, 0, stream>>>(st3, wt[3], bi[3],
                                                       nullptr, fin, 0);

  softmax_thresh<<<BATCH * F_OUT, 256, 0, stream>>>(fin, (float*)d_out);
}
